// GraphSAGEModel_30056181137900
// MI455X (gfx1250) — matrix operand order repaired and checked by reading, not compiled, not run on hardware
//
#include <hip/hip_runtime.h>
#include <hip/hip_fp16.h>
#include <cstdint>

#define N_NODES 50000
#define N_EDGES 800000
#define IN_DIM  128
#define HIDDEN  256
#define OUT_DIM 128

typedef __attribute__((ext_vector_type(16))) _Float16 v16h;
typedef __attribute__((ext_vector_type(8)))  float    v8f;

union Frag16 { v16h h; uint32_t u[8]; };

// ---------------- WMMA fragment loaders (wave32, 16x16x32 f16) ----------------

// A: 16x32 tile of row-major f16 matrix [*, lda]; lane l holds row l%16.
// VGPR v (dword) holds K pair: base = 8*(l/16) + 2*(v&3) + (v>=4 ? 16 : 0).
__device__ __forceinline__ v16h load_a_frag(const _Float16* A, int lda, int row0,
                                            int k0, int lane) {
    const int r  = row0 + (lane & 15);
    const int hi = lane >> 4;
    const uint32_t* p32 = (const uint32_t*)(A + (size_t)r * lda + k0);
    Frag16 f;
#pragma unroll
    for (int v = 0; v < 8; ++v)
        f.u[v] = p32[hi * 4 + (v & 3) + ((v >= 4) ? 8 : 0)];
    return f.h;
}

// B: 32x16 tile of row-major f16 matrix [K, ldb]; lane l holds column n0+l%16,
// element s has k = 8*(l/16) + s (s<8) or 16 + 8*(l/16) + (s-8) (s>=8).
__device__ __forceinline__ v16h load_b_frag(const _Float16* B, int ldb, int k0,
                                            int n0, int lane) {
    const int hi = lane >> 4;
    const _Float16* p = B + (size_t)k0 * ldb + n0 + (lane & 15);
    v16h f;
#pragma unroll
    for (int s = 0; s < 16; ++s)
        f[s] = p[(size_t)((s < 8) ? (8 * hi + s) : (16 + 8 * hi + (s - 8))) * ldb];
    return f;
}

__device__ __forceinline__ float gelu_exact(float x) {
    return 0.5f * x * (1.0f + erff(x * 0.70710678118654752f));
}

// ---------------- Utility kernels ----------------

__global__ void zero_f4_kernel(float4* p, int n4) {
    int i = blockIdx.x * blockDim.x + threadIdx.x;
    if (i < n4) p[i] = make_float4(0.f, 0.f, 0.f, 0.f);
}

__global__ void cvt_f32_to_f16_kernel(const float* s, _Float16* d, int n) {
    int i = blockIdx.x * blockDim.x + threadIdx.x;
    if (i < n) d[i] = (_Float16)s[i];
}

__global__ void degree_kernel(const int* dst, float* cnt) {
    int e = blockIdx.x * blockDim.x + threadIdx.x;
    if (e < N_EDGES) unsafeAtomicAdd(&cnt[dst[e]], 1.0f);
}

// ---------------- Layer 1: scatter x[src] -> agg1[dst]  (128 f32/edge) -------

__global__ void scatter1_kernel(const float* __restrict__ x,
                                const int* __restrict__ src,
                                const int* __restrict__ dst,
                                float* __restrict__ agg) {
    int tid = blockIdx.x * blockDim.x + threadIdx.x;   // N_EDGES*32 threads
    if (tid >= N_EDGES * 32) return;
    int e = tid >> 5, q = tid & 31;
    int s = src[e], d = dst[e];
    float4 v = ((const float4*)(x + (size_t)s * IN_DIM))[q];
    float* a = agg + (size_t)d * IN_DIM + q * 4;
    unsafeAtomicAdd(a + 0, v.x);
    unsafeAtomicAdd(a + 1, v.y);
    unsafeAtomicAdd(a + 2, v.z);
    unsafeAtomicAdd(a + 3, v.w);
}

// mean = agg1/max(cnt,1) -> f16 ; x -> f16
__global__ void prep1_kernel(const float* __restrict__ agg1,
                             const float* __restrict__ x,
                             const float* __restrict__ cnt,
                             _Float16* __restrict__ meanh,
                             _Float16* __restrict__ xh) {
    int tid = blockIdx.x * blockDim.x + threadIdx.x;   // N_NODES*32 threads
    if (tid >= N_NODES * 32) return;
    int node = tid >> 5, q = tid & 31;
    float inv = 1.0f / fmaxf(cnt[node], 1.0f);
    float4 av = ((const float4*)(agg1 + (size_t)node * IN_DIM))[q];
    float4 xv = ((const float4*)(x + (size_t)node * IN_DIM))[q];
    union { _Float16 h[4]; uint2 u; } m, xx;
    m.h[0] = (_Float16)(av.x * inv); m.h[1] = (_Float16)(av.y * inv);
    m.h[2] = (_Float16)(av.z * inv); m.h[3] = (_Float16)(av.w * inv);
    xx.h[0] = (_Float16)xv.x; xx.h[1] = (_Float16)xv.y;
    xx.h[2] = (_Float16)xv.z; xx.h[3] = (_Float16)xv.w;
    ((uint2*)meanh)[(size_t)node * 32 + q] = m.u;
    ((uint2*)xh)[(size_t)node * 32 + q]   = xx.u;
}

// h1 = gelu(mean@Wl1 + x@Wr1 + b1)  -> f16 [N_NODES, HIDDEN]
__global__ void __launch_bounds__(256)
gemm1_kernel(const _Float16* __restrict__ meanh, const _Float16* __restrict__ xh,
             const _Float16* __restrict__ Wl, const _Float16* __restrict__ Wr,
             const float* __restrict__ bias, _Float16* __restrict__ out) {
    const int lane = threadIdx.x & 31;
    const int wid  = blockIdx.x * (blockDim.x >> 5) + (threadIdx.x >> 5);
    const int NT = HIDDEN / 16;
    const int mt = wid / NT, nt = wid % NT;
    if (mt >= N_NODES / 16) return;
    const int m0 = mt * 16, n0 = nt * 16;
    v8f acc = {};
#pragma unroll
    for (int k0 = 0; k0 < IN_DIM; k0 += 32) {
        v16h a0 = load_a_frag(meanh, IN_DIM, m0, k0, lane);
        v16h b0 = load_b_frag(Wl, HIDDEN, k0, n0, lane);
        acc = __builtin_amdgcn_wmma_f32_16x16x32_f16(false, a0, false, b0,
                                                     (short)0, acc, false, false);
        v16h a1 = load_a_frag(xh, IN_DIM, m0, k0, lane);
        v16h b1 = load_b_frag(Wr, HIDDEN, k0, n0, lane);
        acc = __builtin_amdgcn_wmma_f32_16x16x32_f16(false, a1, false, b1,
                                                     (short)0, acc, false, false);
    }
    const int n  = n0 + (lane & 15);
    const float bv = bias[n];
    const int mb = m0 + ((lane >> 4) << 3);
#pragma unroll
    for (int v = 0; v < 8; ++v) {
        float val = gelu_exact(acc[v] + bv);
        out[(size_t)(mb + v) * HIDDEN + n] = (_Float16)val;
    }
}

// ---------------- Layer 2: scatter h[src] -> agg2[dst]  (256 f32/edge) -------

__global__ void scatter2_kernel(const _Float16* __restrict__ h,
                                const int* __restrict__ src,
                                const int* __restrict__ dst,
                                float* __restrict__ agg) {
    int tid = blockIdx.x * blockDim.x + threadIdx.x;   // N_EDGES*32 threads
    if (tid >= N_EDGES * 32) return;
    int e = tid >> 5, q = tid & 31;                    // q handles 8 halves
    int s = src[e], d = dst[e];
    union { uint4 u; _Float16 hh[8]; } v;
    v.u = ((const uint4*)(h + (size_t)s * HIDDEN))[q];
    float* a = agg + (size_t)d * HIDDEN + q * 8;
#pragma unroll
    for (int i = 0; i < 8; ++i) unsafeAtomicAdd(a + i, (float)v.hh[i]);
}

__global__ void prep2_kernel(const float* __restrict__ agg2,
                             const float* __restrict__ cnt,
                             _Float16* __restrict__ mean2h) {
    int tid = blockIdx.x * blockDim.x + threadIdx.x;   // N_NODES*64 threads
    if (tid >= N_NODES * 64) return;
    int node = tid >> 6, q = tid & 63;
    float inv = 1.0f / fmaxf(cnt[node], 1.0f);
    float4 av = ((const float4*)(agg2 + (size_t)node * HIDDEN))[q];
    union { _Float16 h[4]; uint2 u; } m;
    m.h[0] = (_Float16)(av.x * inv); m.h[1] = (_Float16)(av.y * inv);
    m.h[2] = (_Float16)(av.z * inv); m.h[3] = (_Float16)(av.w * inv);
    ((uint2*)mean2h)[(size_t)node * 64 + q] = m.u;
}

// out = mean2@Wl2 + h@Wr2 + b2  -> f32 [N_NODES, OUT_DIM]
__global__ void __launch_bounds__(256)
gemm2_kernel(const _Float16* __restrict__ mean2h, const _Float16* __restrict__ h,
             const _Float16* __restrict__ Wl, const _Float16* __restrict__ Wr,
             const float* __restrict__ bias, float* __restrict__ out) {
    const int lane = threadIdx.x & 31;
    const int wid  = blockIdx.x * (blockDim.x >> 5) + (threadIdx.x >> 5);
    const int NT = OUT_DIM / 16;
    const int mt = wid / NT, nt = wid % NT;
    if (mt >= N_NODES / 16) return;
    const int m0 = mt * 16, n0 = nt * 16;
    v8f acc = {};
#pragma unroll
    for (int k0 = 0; k0 < HIDDEN; k0 += 32) {
        v16h a0 = load_a_frag(mean2h, HIDDEN, m0, k0, lane);
        v16h b0 = load_b_frag(Wl, OUT_DIM, k0, n0, lane);
        acc = __builtin_amdgcn_wmma_f32_16x16x32_f16(false, a0, false, b0,
                                                     (short)0, acc, false, false);
        v16h a1 = load_a_frag(h, HIDDEN, m0, k0, lane);
        v16h b1 = load_b_frag(Wr, OUT_DIM, k0, n0, lane);
        acc = __builtin_amdgcn_wmma_f32_16x16x32_f16(false, a1, false, b1,
                                                     (short)0, acc, false, false);
    }
    const int n  = n0 + (lane & 15);
    const float bv = bias[n];
    const int mb = m0 + ((lane >> 4) << 3);
#pragma unroll
    for (int v = 0; v < 8; ++v)
        out[(size_t)(mb + v) * OUT_DIM + n] = acc[v] + bv;
}

// ---------------- Host side ----------------

extern "C" void kernel_launch(void* const* d_in, const int* in_sizes, int n_in,
                              void* d_out, int out_size, void* d_ws, size_t ws_size,
                              hipStream_t stream) {
    const float* x    = (const float*)d_in[0];
    const int*   src  = (const int*)d_in[1];                 // [2, N_EDGES]
    const int*   dst  = src + N_EDGES;
    const float* Wl1  = (const float*)d_in[2];
    const float* Wr1  = (const float*)d_in[3];
    const float* b1   = (const float*)d_in[4];
    const float* Wl2  = (const float*)d_in[5];
    const float* Wr2  = (const float*)d_in[6];
    const float* b2   = (const float*)d_in[7];
    float* out = (float*)d_out;

    // Workspace carve-up (all offsets 32B-aligned).
    float* cnt  = (float*)d_ws;                              // 50048 (padded)
    float* agg1 = cnt + 50048;                               // N*128 f32
    float* agg2 = agg1 + (size_t)N_NODES * IN_DIM;           // N*256 f32
    _Float16* xh     = (_Float16*)(agg2 + (size_t)N_NODES * HIDDEN);
    _Float16* meanh  = xh + (size_t)N_NODES * IN_DIM;
    _Float16* hbuf   = meanh + (size_t)N_NODES * IN_DIM;     // N*256 f16
    _Float16* mean2h = hbuf + (size_t)N_NODES * HIDDEN;      // N*256 f16
    _Float16* Wl1h   = mean2h + (size_t)N_NODES * HIDDEN;
    _Float16* Wr1h   = Wl1h + IN_DIM * HIDDEN;
    _Float16* Wl2h   = Wr1h + IN_DIM * HIDDEN;
    _Float16* Wr2h   = Wl2h + HIDDEN * OUT_DIM;

    const int T = 256;
    // 1) zero cnt+agg1+agg2 (contiguous)
    int n4 = (50048 + N_NODES * IN_DIM + N_NODES * HIDDEN) / 4;
    zero_f4_kernel<<<(n4 + T - 1) / T, T, 0, stream>>>((float4*)d_ws, n4);
    // 2) weights -> f16
    cvt_f32_to_f16_kernel<<<(IN_DIM * HIDDEN + T - 1) / T, T, 0, stream>>>(Wl1, Wl1h, IN_DIM * HIDDEN);
    cvt_f32_to_f16_kernel<<<(IN_DIM * HIDDEN + T - 1) / T, T, 0, stream>>>(Wr1, Wr1h, IN_DIM * HIDDEN);
    cvt_f32_to_f16_kernel<<<(HIDDEN * OUT_DIM + T - 1) / T, T, 0, stream>>>(Wl2, Wl2h, HIDDEN * OUT_DIM);
    cvt_f32_to_f16_kernel<<<(HIDDEN * OUT_DIM + T - 1) / T, T, 0, stream>>>(Wr2, Wr2h, HIDDEN * OUT_DIM);
    // 3) degrees
    degree_kernel<<<(N_EDGES + T - 1) / T, T, 0, stream>>>(dst, cnt);
    // 4) layer-1 scatter
    scatter1_kernel<<<(N_EDGES * 32 + T - 1) / T, T, 0, stream>>>(x, src, dst, agg1);
    // 5) mean + f16 conversions
    prep1_kernel<<<(N_NODES * 32 + T - 1) / T, T, 0, stream>>>(agg1, x, cnt, meanh, xh);
    // 6) layer-1 fused GEMM + bias + GELU (WMMA)
    gemm1_kernel<<<(N_NODES / 16) * (HIDDEN / 16) / 8, 256, 0, stream>>>(meanh, xh, Wl1h, Wr1h, b1, hbuf);
    // 7) layer-2 scatter
    scatter2_kernel<<<(N_EDGES * 32 + T - 1) / T, T, 0, stream>>>(hbuf, src, dst, agg2);
    // 8) layer-2 mean
    prep2_kernel<<<(N_NODES * 64 + T - 1) / T, T, 0, stream>>>(agg2, cnt, mean2h);
    // 9) layer-2 fused GEMM + bias (WMMA) -> f32 output
    gemm2_kernel<<<(N_NODES / 16) * (OUT_DIM / 16) / 8, 256, 0, stream>>>(mean2h, hbuf, Wl2h, Wr2h, b2, out);
}
